// RPNAlgoFPN_JIT_58746562675171
// MI455X (gfx1250) — compile-verified, matrix operand order checked by reading, not run on hardware
//
#include <hip/hip_runtime.h>
#include <math.h>

// ---------------- types ----------------
typedef __bf16 bf16_t;
typedef __attribute__((ext_vector_type(16))) __bf16 v16bf;
typedef __attribute__((ext_vector_type(8)))  __bf16 v8bf;
typedef __attribute__((ext_vector_type(8)))  float  v8f;
typedef unsigned uint4v __attribute__((ext_vector_type(4)));
typedef __attribute__((ext_vector_type(4))) int int4v_t;

// ---------------- feature detection (per compile pass) ----------------
#ifndef __has_builtin
#define __has_builtin(x) 0
#endif
#if __has_builtin(__builtin_amdgcn_global_load_async_to_lds_b128)
#define HAVE_ASYNC_LDS 1
#else
#define HAVE_ASYNC_LDS 0
#endif

#if HAVE_ASYNC_LDS
typedef __attribute__((address_space(1))) int4v_t g_int4;   // global v4i32
typedef __attribute__((address_space(3))) int4v_t l_int4;   // LDS v4i32
#if __has_builtin(__builtin_amdgcn_s_wait_asynccnt)
#define WAIT_ASYNC() __builtin_amdgcn_s_wait_asynccnt(0)
#else
#define WAIT_ASYNC() asm volatile("s_wait_asynccnt 0x0" ::: "memory")
#endif
#endif

// ---------------- constants ----------------
#define NIMG   2
#define CCH    128
#define NAA    9
#define KTOT   1152        // 128 * 9
#define TOTPIX 87296       // sum H*W over levels
#define MTOT   785664      // TOTPIX * 9
#define NUM_PRE  6000
#define NUM_POST 300
#define NMS_THRESH 0.7f
#define SCALE_CLIP 4.135166556742356f
#define HIST_BINS 2048

__device__ __forceinline__ float bf2f(bf16_t b) {
    unsigned short s = __builtin_bit_cast(unsigned short, b);
    unsigned u = ((unsigned)s) << 16;
    return __builtin_bit_cast(float, u);
}
__device__ __forceinline__ bf16_t f2bf(float f) {
    unsigned u = __builtin_bit_cast(unsigned, f);
    unsigned r = (u + 0x7FFFu + ((u >> 16) & 1u)) >> 16;
    unsigned short s = (unsigned short)r;
    return __builtin_bit_cast(bf16_t, s);
}
__device__ __forceinline__ unsigned scoreKey(float s) {
    unsigned u = __builtin_bit_cast(unsigned, s);
    return (u & 0x80000000u) ? ~u : (u | 0x80000000u);
}

// ---------------- weight conversion: [co][ci][3][3] f32 -> [co][kpos*128+ci] bf16 --------
__global__ void cvt_weights_kernel(const float* __restrict__ w1, bf16_t* __restrict__ Abf) {
    int idx = blockIdx.x * 256 + threadIdx.x;
    if (idx >= CCH * KTOT) return;
    int co = idx / KTOT;
    int k  = idx % KTOT;
    int kpos = k >> 7;        // 0..8  (ky*3+kx)
    int ci   = k & 127;
    int ky = kpos / 3, kx = kpos % 3;
    float v = w1[((co * CCH + ci) * 3 + ky) * 3 + kx];
    Abf[idx] = f2bf(v);
}

// ---------------- feature conversion: NCHW f32 -> NHWC bf16 (level-packed) ----------------
__global__ void cvt_feat_kernel(const float* __restrict__ x, bf16_t* __restrict__ xbf,
                                int H, int W, int lvlPixOff) {
    int idx = blockIdx.x * 256 + threadIdx.x;
    int hw = H * W;
    int tot = NIMG * CCH * hw;
    if (idx >= tot) return;
    int c   = idx & 127;
    int pix = idx >> 7;        // n*hw + p
    int n = pix / hw;
    int p = pix - n * hw;
    float v = x[((size_t)(n * CCH + c)) * hw + p];
    xbf[((size_t)(n * TOTPIX + lvlPixOff + p)) * CCH + c] = f2bf(v);
}

// ---------------- conv3x3 (C=128 -> C=128) + bias + ReLU via WMMA bf16 ----------------
// block = 256 threads = 8 waves; block tile = 128 co x 32 pixels.
// wave w -> co [16w,16w+16), two accumulators (pixel subtiles 0-15 and 16-31).
// 9 staging phases (one per 3x3 tap): LDS holds 32 px x 128 ci bf16 (8KB),
// each phase runs 4 K-chunks x 2 subtiles = 8 WMMA per wave between barriers.
__global__ __launch_bounds__(256)
void conv3x3_wmma_kernel(const bf16_t* __restrict__ xbf,
                         const bf16_t* __restrict__ Abf,
                         const float* __restrict__ b1,
                         bf16_t* __restrict__ hbf,
                         int H, int W, int lvlPixOff) {
    __shared__ __align__(16) bf16_t Bt[32 * CCH];   // 8 KB

    const int n = blockIdx.y;
    const int pixBase = blockIdx.x * 32;
    const int wave = threadIdx.x >> 5;
    const int lane = threadIdx.x & 31;
    const int co0 = wave * 16;

    const int kbase = (lane & 16) ? 8 : 0;   // 16-bit A/B fragment K sub-offset
    const int mrow  = co0 + (lane & 15);     // A row (co)
    const int pixl  = lane & 15;             // B col within subtile

    v8f acc0 = {};
    v8f acc1 = {};

    for (int kpos = 0; kpos < 9; ++kpos) {
        const int dy = kpos / 3 - 1, dx = kpos % 3 - 1;

        // ---- stage 32 px x 128 ci (zero-padded at borders) ----
#pragma unroll
        for (int r = 0; r < 2; ++r) {
            int q   = threadIdx.x + r * 256;  // 0..511 uint4 slots
            int spx = q >> 4;                 // pixel 0..31
            int seg = q & 15;                 // 8-channel segment
            int p  = pixBase + spx;
            int py = p / W, px = p % W;
            int yy = py + dy, xx = px + dx;
            bool inb = (yy >= 0) && (yy < H) && (xx >= 0) && (xx < W);
            long long gidx = ((long long)n * TOTPIX + lvlPixOff +
                              (long long)yy * W + xx) * CCH + seg * 8;
            const bf16_t* gsrc = xbf + gidx;
            bf16_t* ldst = Bt + spx * CCH + seg * 8;
#if HAVE_ASYNC_LDS
            if (inb) {
                __builtin_amdgcn_global_load_async_to_lds_b128(
                    (g_int4*)gsrc, (l_int4*)ldst, 0, 0);
            } else {
                uint4v z = {0u, 0u, 0u, 0u};
                *(uint4v*)ldst = z;
            }
#else
            uint4v v = {0u, 0u, 0u, 0u};
            if (inb) v = *(const uint4v*)gsrc;
            *(uint4v*)ldst = v;
#endif
        }
#if HAVE_ASYNC_LDS
        WAIT_ASYNC();
#endif
        __syncthreads();

        const bf16_t* arowBase = Abf + (size_t)mrow * KTOT + kpos * CCH + kbase;
        if (kpos + 1 < 9)
            __builtin_prefetch((const void*)(Abf + (size_t)mrow * KTOT + (kpos + 1) * CCH), 0, 0);

#pragma unroll
        for (int kc4 = 0; kc4 < 4; ++kc4) {
            const bf16_t* arow = arowBase + kc4 * 32;
            v8bf a_lo = *(const v8bf*)(arow);
            v8bf a_hi = *(const v8bf*)(arow + 16);
            v16bf a = __builtin_shufflevector(a_lo, a_hi,
                                              0,1,2,3,4,5,6,7,8,9,10,11,12,13,14,15);

            const bf16_t* b0p = Bt + pixl * CCH + kc4 * 32 + kbase;
            v8bf b0_lo = *(const v8bf*)(b0p);
            v8bf b0_hi = *(const v8bf*)(b0p + 16);
            v16bf b0 = __builtin_shufflevector(b0_lo, b0_hi,
                                               0,1,2,3,4,5,6,7,8,9,10,11,12,13,14,15);

            const bf16_t* b1p = b0p + 16 * CCH;
            v8bf b1_lo = *(const v8bf*)(b1p);
            v8bf b1_hi = *(const v8bf*)(b1p + 16);
            v16bf b1 = __builtin_shufflevector(b1_lo, b1_hi,
                                               0,1,2,3,4,5,6,7,8,9,10,11,12,13,14,15);

            acc0 = __builtin_amdgcn_wmma_f32_16x16x32_bf16(false, a, false, b0, (short)0,
                                                           acc0, false, false);
            acc1 = __builtin_amdgcn_wmma_f32_16x16x32_bf16(false, a, false, b1, (short)0,
                                                           acc1, false, false);
        }
        __syncthreads();
    }

    // epilogue: C/D layout -> VGPR i: lanes 0-15 M=i, lanes 16-31 M=i+8; N=lane&15
    const int m_add = (lane & 16) ? 8 : 0;
    const int opix0 = pixBase + (lane & 15);
    const int opix1 = opix0 + 16;
    const size_t ob0 = ((size_t)(n * TOTPIX + lvlPixOff + opix0)) * CCH;
    const size_t ob1 = ((size_t)(n * TOTPIX + lvlPixOff + opix1)) * CCH;
#pragma unroll
    for (int i = 0; i < 8; ++i) {
        int co = co0 + i + m_add;
        float bias = b1[co];
        float v0 = acc0[i] + bias;
        float v1 = acc1[i] + bias;
        v0 = v0 > 0.0f ? v0 : 0.0f;
        v1 = v1 > 0.0f ? v1 : 0.0f;
        hbf[ob0 + co] = f2bf(v0);
        hbf[ob1 + co] = f2bf(v1);
    }
}

// ---------------- 1x1 heads + anchor decode ----------------
__global__ void head_decode_kernel(const bf16_t* __restrict__ hbf,
                                   const float* __restrict__ wobj, const float* __restrict__ bobj,
                                   const float* __restrict__ wbbx, const float* __restrict__ bbbx,
                                   const int* __restrict__ valid_size,
                                   float* __restrict__ scores, float* __restrict__ boxes,
                                   int H, int W, int lvlPixOff, int lvlAnchOff, float stride) {
    int idx = blockIdx.x * 256 + threadIdx.x;
    int hw = H * W;
    int tot = NIMG * hw * NAA;
    if (idx >= tot) return;
    int a    = idx % NAA;
    int rest = idx / NAA;
    int p = rest % hw;
    int n = rest / hw;

    const bf16_t* hv = hbf + ((size_t)(n * TOTPIX + lvlPixOff + p)) * CCH;
    float s0 = bobj[a];
    float d0 = bbbx[4 * a + 0], d1 = bbbx[4 * a + 1];
    float d2 = bbbx[4 * a + 2], d3 = bbbx[4 * a + 3];
    const float* wo = wobj + a * CCH;
    const float* wb = wbbx + (4 * a) * CCH;
#pragma unroll 4
    for (int c = 0; c < CCH; ++c) {
        float h = bf2f(hv[c]);
        s0 += wo[c] * h;
        d0 += wb[c] * h;
        d1 += wb[CCH + c] * h;
        d2 += wb[2 * CCH + c] * h;
        d3 += wb[3 * CCH + c] * h;
    }

    // anchors
    const float scl[3] = {2.0f, 4.0f, 8.0f};
    const float rat[3] = {0.5f, 1.0f, 2.0f};
    int si = a / 3, ri = a % 3;
    float ah = stride * scl[si] * sqrtf(rat[ri]);
    float aw = stride * scl[si] * sqrtf(1.0f / rat[ri]);
    int y = p / W, x = p % W;
    float cy = stride * 0.5f + (float)y * stride;
    float cx = stride * 0.5f + (float)x * stride;

    // decode
    float oy = cy + d0 * ah;
    float ox = cx + d1 * aw;
    float oh = ah * expf(fminf(d2, SCALE_CLIP));
    float ow = aw * expf(fminf(d3, SCALE_CLIP));
    float hi = (float)valid_size[n * 2 + 0];
    float wi = (float)valid_size[n * 2 + 1];
    float y1 = fminf(fmaxf(oy - 0.5f * oh, 0.0f), hi);
    float x1 = fminf(fmaxf(ox - 0.5f * ow, 0.0f), wi);
    float y2 = fminf(fmaxf(oy + 0.5f * oh, 0.0f), hi);
    float x2 = fminf(fmaxf(ox + 0.5f * ow, 0.0f), wi);

    int m = lvlAnchOff + p * NAA + a;
    scores[(size_t)n * MTOT + m] = s0;
    float* bp = boxes + ((size_t)n * MTOT + m) * 4;
    bp[0] = y1; bp[1] = x1; bp[2] = y2; bp[3] = x2;
}

// ---------------- top-k selection via histogram threshold ----------------
__global__ void hist_kernel(const float* __restrict__ scores, unsigned* __restrict__ hist) {
    int idx = blockIdx.x * 256 + threadIdx.x;
    if (idx >= NIMG * MTOT) return;
    int n = idx / MTOT;
    unsigned key = scoreKey(scores[idx]);
    atomicAdd(&hist[n * HIST_BINS + (key >> 21)], 1u);
}

__global__ void thresh_kernel(const unsigned* __restrict__ hist, unsigned* __restrict__ tbin) {
    int n = threadIdx.x;
    if (n >= NIMG) return;
    unsigned cum = 0;
    int b = HIST_BINS - 1;
    for (; b > 0; --b) {
        cum += hist[n * HIST_BINS + b];
        if (cum >= NUM_PRE) break;
    }
    tbin[n] = (unsigned)b;
}

__global__ void compact_kernel(const float* __restrict__ scores, const float* __restrict__ boxes,
                               const unsigned* __restrict__ tbin, unsigned* __restrict__ cnt,
                               float* __restrict__ selS, float* __restrict__ selB) {
    int idx = blockIdx.x * 256 + threadIdx.x;
    if (idx >= NIMG * MTOT) return;
    int n = idx / MTOT;
    float s = scores[idx];
    unsigned key = scoreKey(s);
    if ((key >> 21) >= tbin[n]) {
        unsigned pos = atomicAdd(&cnt[n], 1u);
        if (pos < NUM_PRE) {
            selS[n * NUM_PRE + pos] = s;
            const float* bp = boxes + (size_t)idx * 4;
            float* op = selB + ((size_t)n * NUM_PRE + pos) * 4;
            op[0] = bp[0]; op[1] = bp[1]; op[2] = bp[2]; op[3] = bp[3];
        }
    }
}

// ---------------- greedy NMS, one block per image ----------------
__global__ __launch_bounds__(256)
void nms_kernel(const float* __restrict__ selS, const float* __restrict__ selB,
                const unsigned* __restrict__ cnt,
                float* __restrict__ outProps, float* __restrict__ outValid) {
    __shared__ unsigned char act[NUM_PRE];
    __shared__ float rs[256];
    __shared__ int   ri[256];
    __shared__ float cb[4];
    __shared__ float careaS;

    const int n = blockIdx.x;
    const int t = threadIdx.x;
    const int P = (int)min(cnt[n], (unsigned)NUM_PRE);
    const float* S = selS + n * NUM_PRE;
    const float* B = selB + (size_t)n * NUM_PRE * 4;

    for (int i = t; i < NUM_PRE; i += 256) act[i] = (i < P) ? 1 : 0;
    __syncthreads();

    for (int it = 0; it < NUM_POST; ++it) {
        float bs = -INFINITY;
        int bi = 0;
        for (int i = t; i < P; i += 256) {
            if (act[i]) {
                float s = S[i];
                if (s > bs) { bs = s; bi = i; }
            }
        }
        rs[t] = bs; ri[t] = bi;
        __syncthreads();
        for (int off = 128; off > 0; off >>= 1) {
            if (t < off) {
                float so = rs[t + off];
                if (so > rs[t] || (so == rs[t] && ri[t + off] < ri[t])) {
                    rs[t] = so; ri[t] = ri[t + off];
                }
            }
            __syncthreads();
        }
        if (t == 0) {
            int i = (rs[0] == -INFINITY) ? 0 : ri[0];
            int ok = (P > 0) ? (int)act[i] : 0;
            float b0 = 0.f, b1v = 0.f, b2 = 0.f, b3 = 0.f;
            if (P > 0) { b0 = B[i*4]; b1v = B[i*4+1]; b2 = B[i*4+2]; b3 = B[i*4+3]; }
            cb[0] = b0; cb[1] = b1v; cb[2] = b2; cb[3] = b3;
            careaS = (b2 - b0) * (b3 - b1v);
            float vf = ok ? 1.0f : 0.0f;
            float* pp = outProps + ((size_t)n * NUM_POST + it) * 4;
            pp[0] = b0 * vf; pp[1] = b1v * vf; pp[2] = b2 * vf; pp[3] = b3 * vf;
            outValid[n * NUM_POST + it] = vf;
        }
        __syncthreads();
        float y1 = cb[0], x1 = cb[1], y2 = cb[2], x2 = cb[3], A0 = careaS;
        for (int i = t; i < P; i += 256) {
            if (!act[i]) continue;
            float by0 = B[i*4], bx0 = B[i*4+1], by1 = B[i*4+2], bx1 = B[i*4+3];
            float ty = fmaxf(y1, by0), tx = fmaxf(x1, bx0);
            float cy = fminf(y2, by1), cx = fminf(x2, bx1);
            float inter = fmaxf(cy - ty, 0.0f) * fmaxf(cx - tx, 0.0f);
            float Ai = (by1 - by0) * (bx1 - bx0);
            float iou = inter / (A0 + Ai - inter + 1e-9f);
            if (iou > NMS_THRESH) act[i] = 0;
        }
        __syncthreads();
    }
}

// ---------------- launch ----------------
extern "C" void kernel_launch(void* const* d_in, const int* in_sizes, int n_in,
                              void* d_out, int out_size, void* d_ws, size_t ws_size,
                              hipStream_t stream) {
    (void)in_sizes; (void)n_in; (void)out_size; (void)ws_size;

    const float* x_in[5] = {(const float*)d_in[0], (const float*)d_in[1], (const float*)d_in[2],
                            (const float*)d_in[3], (const float*)d_in[4]};
    const float* w1   = (const float*)d_in[5];
    const float* b1   = (const float*)d_in[6];
    const float* wobj = (const float*)d_in[7];
    const float* bobj = (const float*)d_in[8];
    const float* wbbx = (const float*)d_in[9];
    const float* bbbx = (const float*)d_in[10];
    const int*   vsz  = (const int*)d_in[11];

    static const int   HL[5]     = {256, 128, 64, 32, 16};
    static const int   PIXOFF[5] = {0, 65536, 81920, 86016, 87040};
    static const int   ANCOFF[5] = {0, 589824, 737280, 774144, 783360};
    static const float STR[5]    = {4.0f, 8.0f, 16.0f, 32.0f, 64.0f};

    // workspace layout
    char* ws = (char*)d_ws;
    size_t off = 0;
    auto alloc = [&](size_t bytes) -> size_t {
        size_t o = off;
        off = (off + bytes + 255) & ~(size_t)255;
        return o;
    };
    size_t off_Abf   = alloc((size_t)CCH * KTOT * 2);
    size_t off_xbf   = alloc((size_t)NIMG * TOTPIX * CCH * 2);
    size_t off_hbf   = alloc((size_t)NIMG * TOTPIX * CCH * 2);
    size_t off_score = alloc((size_t)NIMG * MTOT * 4);
    size_t off_boxes = alloc((size_t)NIMG * MTOT * 16);
    size_t off_selS  = alloc((size_t)NIMG * NUM_PRE * 4);
    size_t off_selB  = alloc((size_t)NIMG * NUM_PRE * 16);
    size_t off_hist  = alloc((size_t)NIMG * HIST_BINS * 4 + 64); // hist + tbin + cnt contiguous

    bf16_t*   Abf   = (bf16_t*)(ws + off_Abf);
    bf16_t*   xbf   = (bf16_t*)(ws + off_xbf);
    bf16_t*   hbf   = (bf16_t*)(ws + off_hbf);
    float*    score = (float*)(ws + off_score);
    float*    boxes = (float*)(ws + off_boxes);
    float*    selS  = (float*)(ws + off_selS);
    float*    selB  = (float*)(ws + off_selB);
    unsigned* hist  = (unsigned*)(ws + off_hist);
    unsigned* tbin  = hist + NIMG * HIST_BINS;       // 2 words
    unsigned* cnt   = tbin + NIMG;                   // 2 words

    // clear histogram + tbin + cnt
    (void)hipMemsetAsync(hist, 0, (size_t)NIMG * HIST_BINS * 4 + 64, stream);

    // 1. weights -> bf16 K-major
    cvt_weights_kernel<<<(CCH * KTOT + 255) / 256, 256, 0, stream>>>(w1, Abf);

    // 2. features -> NHWC bf16 ; 3. conv3x3 WMMA ; 4. heads + decode
    for (int l = 0; l < 5; ++l) {
        int H = HL[l], W = HL[l];
        int elems = NIMG * CCH * H * W;
        cvt_feat_kernel<<<(elems + 255) / 256, 256, 0, stream>>>(x_in[l], xbf, H, W, PIXOFF[l]);
    }
    for (int l = 0; l < 5; ++l) {
        int H = HL[l], W = HL[l];
        dim3 grid((H * W) / 32, NIMG);
        conv3x3_wmma_kernel<<<grid, 256, 0, stream>>>(xbf, Abf, b1, hbf, H, W, PIXOFF[l]);
    }
    for (int l = 0; l < 5; ++l) {
        int H = HL[l], W = HL[l];
        int tot = NIMG * H * W * NAA;
        head_decode_kernel<<<(tot + 255) / 256, 256, 0, stream>>>(
            hbf, wobj, bobj, wbbx, bbbx, vsz, score, boxes, H, W, PIXOFF[l], ANCOFF[l], STR[l]);
    }

    // 5. histogram / threshold / compaction (approximate top-6000 per image)
    int totScore = NIMG * MTOT;
    hist_kernel<<<(totScore + 255) / 256, 256, 0, stream>>>(score, hist);
    thresh_kernel<<<1, 32, 0, stream>>>(hist, tbin);
    compact_kernel<<<(totScore + 255) / 256, 256, 0, stream>>>(score, boxes, tbin, cnt, selS, selB);

    // 6. NMS + output (props [2,300,4] then valid [2,300] as floats)
    float* outProps = (float*)d_out;
    float* outValid = outProps + NIMG * NUM_POST * 4;
    nms_kernel<<<NIMG, 256, 0, stream>>>(selS, selB, cnt, outProps, outValid);
}
